// MultiHeadAttention_35802847379830
// MI455X (gfx1250) — matrix operand order repaired and checked by reading, not compiled, not run on hardware
//
#include <hip/hip_runtime.h>
#include <hip/hip_bf16.h>

// ---------------------------------------------------------------------------
// MHA forward for MI455X (gfx1250, wave32, WMMA).
//   B=1, S=4096, D=768, H=12, DK=64
// All matmuls via v_wmma_f32_16x16x32_bf16 (fp32 accum). GEMM tiles staged
// with async global->LDS copies (ASYNCcnt) and double-buffered LDS.
// ---------------------------------------------------------------------------

#define S_LEN 4096
#define D_DIM 768
#define H_NUM 12
#define DK_DIM 64

typedef __bf16 bf16_t;
typedef __attribute__((ext_vector_type(16))) __bf16 v16bf;
typedef __attribute__((ext_vector_type(8)))  __bf16 v8bf;
typedef __attribute__((ext_vector_type(4)))  __bf16 v4bf;
typedef __attribute__((ext_vector_type(8)))  float  v8f;
typedef __attribute__((ext_vector_type(4)))  int    v4i;

#define LDS_STRIDE 40   // 128x32 tile rows padded to 40 bf16 (80B, 16B-aligned)

#if __has_builtin(__builtin_amdgcn_global_load_async_to_lds_b128)
#define HAVE_ASYNC_LDS 1
#else
#define HAVE_ASYNC_LDS 0
#endif

#define AS_GLOBAL __attribute__((address_space(1)))
#define AS_LDS    __attribute__((address_space(3)))

// ---------------------------------------------------------------------------
// fp32 -> bf16 conversion (vectorized x4)
// ---------------------------------------------------------------------------
__global__ void cvt_f32_to_bf16(const float* __restrict__ src,
                                bf16_t* __restrict__ dst, int n4) {
    int i = blockIdx.x * blockDim.x + threadIdx.x;
    if (i >= n4) return;
    float4 f = ((const float4*)src)[i];
    v4bf o;
    o.x = (bf16_t)f.x; o.y = (bf16_t)f.y; o.z = (bf16_t)f.z; o.w = (bf16_t)f.w;
    ((v4bf*)dst)[i] = o;
}

// ---------------------------------------------------------------------------
// Vt[h][d][s] = Yv[s][h*64+d]   (so PV can be A*B^T like every other GEMM)
// ---------------------------------------------------------------------------
__global__ void transpose_v(const bf16_t* __restrict__ Yv,
                            bf16_t* __restrict__ Vt) {
    int idx = blockIdx.x * blockDim.x + threadIdx.x;   // [hd][s], s fastest
    int s  = idx & (S_LEN - 1);
    int hd = idx >> 12;
    if (hd < D_DIM)
        Vt[(long)hd * S_LEN + s] = Yv[(long)s * D_DIM + hd];
}

// ---------------------------------------------------------------------------
// Generic bf16 WMMA GEMM:  C[z] = scale * A[z](MxK) * B[z](NxK)^T + bias
// Block tile 128x128, 8 waves (wave32), wave tile 64x32 = 4x2 wmma frags.
// Double-buffered LDS tiles; async global->LDS staging when available.
// ---------------------------------------------------------------------------
__global__ __launch_bounds__(256, 2)
void gemm_bf16_wmma(const bf16_t* __restrict__ A, long lda, long sAz,
                    const bf16_t* __restrict__ B, long ldb, long sBz,
                    const float* __restrict__ bias,
                    bf16_t* __restrict__ Cb, float* __restrict__ Cf,
                    long ldc, long sCz,
                    int M, int N, int K, float scale) {
    __shared__ bf16_t As[2][128 * LDS_STRIDE];
    __shared__ bf16_t Bs[2][128 * LDS_STRIDE];

    const int z = blockIdx.z;
    A += (long)z * sAz;
    B += (long)z * sBz;

    const int tid  = threadIdx.x;
    const int lane = tid & 31;
    const int wave = tid >> 5;
    const int wm   = wave & 1;    // 2 x 64-row slices
    const int wn   = wave >> 1;   // 4 x 32-col slices
    const int blockM = blockIdx.y * 128;
    const int blockN = blockIdx.x * 128;

    // cooperative tile-load indices: each thread moves 32B (16 bf16)
    const int lr = tid >> 1;            // row 0..127 within tile
    const int lh = (tid & 1) * 16;      // element offset 0 or 16

    // fragment assembly indices (contiguous K order, same for A and B)
    const int laneK8  = (lane >> 4) * 8;    // unused; A uses laneK16 like B
    const int laneK16 = (lane >> 4) * 16;   // B: lanes16-31 hold K=16..31
    const int lm      = lane & 15;

    const bf16_t bz = (bf16_t)0.0f;
    const v8bf vbz = { bz, bz, bz, bz, bz, bz, bz, bz };
    const v8f  fz = { 0.f, 0.f, 0.f, 0.f, 0.f, 0.f, 0.f, 0.f };
    v8f acc[4][2];
#pragma unroll
    for (int i = 0; i < 4; ++i)
#pragma unroll
        for (int j = 0; j < 2; ++j) acc[i][j] = fz;

    // stage one 128x32 K-tile of A and B into LDS buffer b
    auto stage = [&](int kt, int b) {
        // ---- A tile ----
        {
            int ar = blockM + lr;
            bf16_t* la = &As[b][lr * LDS_STRIDE + lh];
            if (ar < M) {
                const bf16_t* ap = A + (long)ar * lda + kt + lh;
#if HAVE_ASYNC_LDS
                __builtin_amdgcn_global_load_async_to_lds_b128(
                    (AS_GLOBAL v4i*)ap, (AS_LDS v4i*)la, 0, 0);
                __builtin_amdgcn_global_load_async_to_lds_b128(
                    (AS_GLOBAL v4i*)(ap + 8), (AS_LDS v4i*)(la + 8), 0, 0);
#else
                *(v8bf*)la       = *(const v8bf*)ap;
                *(v8bf*)(la + 8) = *(const v8bf*)(ap + 8);
#endif
            } else {
                *(v8bf*)la       = vbz;
                *(v8bf*)(la + 8) = vbz;
            }
        }
        // ---- B tile (rows are N; B is N x K) ----
        {
            int bn = blockN + lr;
            bf16_t* lb = &Bs[b][lr * LDS_STRIDE + lh];
            if (bn < N) {
                const bf16_t* bp = B + (long)bn * ldb + kt + lh;
#if HAVE_ASYNC_LDS
                __builtin_amdgcn_global_load_async_to_lds_b128(
                    (AS_GLOBAL v4i*)bp, (AS_LDS v4i*)lb, 0, 0);
                __builtin_amdgcn_global_load_async_to_lds_b128(
                    (AS_GLOBAL v4i*)(bp + 8), (AS_LDS v4i*)(lb + 8), 0, 0);
#else
                *(v8bf*)lb       = *(const v8bf*)bp;
                *(v8bf*)(lb + 8) = *(const v8bf*)(bp + 8);
#endif
            } else {
                *(v8bf*)lb       = vbz;
                *(v8bf*)(lb + 8) = vbz;
            }
        }
    };

    // prologue: fill buffer 0
    stage(0, 0);
#if HAVE_ASYNC_LDS
    asm volatile("s_wait_asynccnt 0x0" ::: "memory");
#endif
    __syncthreads();

    int buf = 0;
    for (int kt = 0; kt < K; kt += 32) {
        const bool more = (kt + 32 < K);
        if (more) stage(kt + 32, buf ^ 1);   // prefetch next tile (other buffer)

        // ---- B fragments: lane holds col n=lm, elems = K laneK16..+15 ----
        v16bf bfrag[2];
#pragma unroll
        for (int nt = 0; nt < 2; ++nt) {
            int n = wn * 32 + nt * 16 + lm;
            v8bf lo = *(const v8bf*)&Bs[buf][n * LDS_STRIDE + laneK16];
            v8bf hi = *(const v8bf*)&Bs[buf][n * LDS_STRIDE + laneK16 + 8];
            bfrag[nt] = __builtin_shufflevector(lo, hi,
                0, 1, 2, 3, 4, 5, 6, 7, 8, 9, 10, 11, 12, 13, 14, 15);
        }
        // ---- A fragments + WMMA ----
#pragma unroll
        for (int mt = 0; mt < 4; ++mt) {
            int m = wm * 64 + mt * 16 + lm;
            // elems 0..15 : K = laneK16..laneK16+15
            v8bf lo = *(const v8bf*)&As[buf][m * LDS_STRIDE + laneK16];
            v8bf hi = *(const v8bf*)&As[buf][m * LDS_STRIDE + laneK16 + 8];
            v16bf afrag = __builtin_shufflevector(lo, hi,
                0, 1, 2, 3, 4, 5, 6, 7, 8, 9, 10, 11, 12, 13, 14, 15);
#pragma unroll
            for (int nt = 0; nt < 2; ++nt) {
                acc[mt][nt] = __builtin_amdgcn_wmma_f32_16x16x32_bf16(
                    false, afrag, false, bfrag[nt],
                    (short)0, acc[mt][nt], false, false);
            }
        }

        if (more) {
#if HAVE_ASYNC_LDS
            asm volatile("s_wait_asynccnt 0x0" ::: "memory");
#endif
            __syncthreads();
        }
        buf ^= 1;
    }

    // ---- store: C/D layout — VGPR r holds (M = r + (lane>=16)*8, N = lm) ----
#pragma unroll
    for (int mt = 0; mt < 4; ++mt) {
        int mrow = blockM + wm * 64 + mt * 16 + (lane >> 4) * 8;
        if (mrow >= M) continue;
#pragma unroll
        for (int nt = 0; nt < 2; ++nt) {
            int ncol = blockN + wn * 32 + nt * 16 + lm;
            if (ncol >= N) continue;
            float bv = bias ? bias[ncol] : 0.0f;
#pragma unroll
            for (int r = 0; r < 8; ++r) {
                float v = acc[mt][nt][r] * scale + bv;
                long off = (long)z * sCz + (long)(mrow + r) * ldc + ncol;
                if (Cf) Cf[off] = v;
                if (Cb) Cb[off] = (bf16_t)v;
            }
        }
    }
}

// ---------------------------------------------------------------------------
// Row softmax over the attention matrix (in place, fp32) + bf16 copy for PV.
// One 256-thread block per row (H*S rows of length S).
// ---------------------------------------------------------------------------
__global__ __launch_bounds__(256)
void softmax_rows(float* __restrict__ attn, bf16_t* __restrict__ pbf) {
    __shared__ float red[256];
    const long row = blockIdx.x;
    float*  r  = attn + row * (long)S_LEN;
    bf16_t* pb = pbf  + row * (long)S_LEN;
    const int t = threadIdx.x;

    float m = -INFINITY;
    for (int c = t; c < S_LEN; c += 256) m = fmaxf(m, r[c]);
    red[t] = m; __syncthreads();
    for (int off = 128; off > 0; off >>= 1) {
        if (t < off) red[t] = fmaxf(red[t], red[t + off]);
        __syncthreads();
    }
    m = red[0]; __syncthreads();

    float sum = 0.f;
    for (int c = t; c < S_LEN; c += 256) {
        float e = __expf(r[c] - m);
        r[c] = e;
        sum += e;
    }
    red[t] = sum; __syncthreads();
    for (int off = 128; off > 0; off >>= 1) {
        if (t < off) red[t] += red[t + off];
        __syncthreads();
    }
    const float inv = 1.0f / red[0];

    for (int c = t; c < S_LEN; c += 256) {
        float p = r[c] * inv;
        r[c]  = p;
        pb[c] = (bf16_t)p;
    }
}

// ---------------------------------------------------------------------------
// host-side launch
// ---------------------------------------------------------------------------
extern "C" void kernel_launch(void* const* d_in, const int* in_sizes, int n_in,
                              void* d_out, int out_size, void* d_ws, size_t ws_size,
                              hipStream_t stream) {
    (void)in_sizes; (void)n_in; (void)out_size; (void)ws_size;

    const float* Q  = (const float*)d_in[0];
    const float* K  = (const float*)d_in[1];
    const float* V  = (const float*)d_in[2];
    const float* Wq = (const float*)d_in[3];
    const float* bq = (const float*)d_in[4];
    const float* Wk = (const float*)d_in[5];
    const float* bk = (const float*)d_in[6];
    const float* Wv = (const float*)d_in[7];
    const float* bv = (const float*)d_in[8];
    const float* Wo = (const float*)d_in[9];
    const float* bo = (const float*)d_in[10];

    float* out  = (float*)d_out;                         // [S, D]
    float* attn = (float*)d_out + (size_t)S_LEN * D_DIM; // [H, S, S]

    const size_t SD = (size_t)S_LEN * D_DIM;
    const size_t DD = (size_t)D_DIM * D_DIM;
    const size_t HSS = (size_t)H_NUM * S_LEN * S_LEN;

    // workspace carve-out (256B aligned)
    size_t o = 0;
    auto alloc = [&](size_t bytes) {
        size_t r = o; o = (o + bytes + 255) & ~(size_t)255; return r;
    };
    char* ws = (char*)d_ws;
    bf16_t* Qbf  = (bf16_t*)(ws + alloc(SD * 2));
    bf16_t* Kbf  = (bf16_t*)(ws + alloc(SD * 2));
    bf16_t* Vbf  = (bf16_t*)(ws + alloc(SD * 2));
    bf16_t* Wqb  = (bf16_t*)(ws + alloc(DD * 2));
    bf16_t* Wkb  = (bf16_t*)(ws + alloc(DD * 2));
    bf16_t* Wvb  = (bf16_t*)(ws + alloc(DD * 2));
    bf16_t* Wob  = (bf16_t*)(ws + alloc(DD * 2));
    bf16_t* Yq   = (bf16_t*)(ws + alloc(SD * 2));
    bf16_t* Yk   = (bf16_t*)(ws + alloc(SD * 2));
    bf16_t* Yv   = (bf16_t*)(ws + alloc(SD * 2));
    bf16_t* Vt   = (bf16_t*)(ws + alloc(SD * 2));   // [H*DK, S]
    bf16_t* Ctx  = (bf16_t*)(ws + alloc(SD * 2));   // [S, D]
    bf16_t* Pbf  = (bf16_t*)(ws + alloc(HSS * 2));  // [H, S, S]

    // 1) conversions
    {
        int n4 = (int)(SD / 4), g = (n4 + 255) / 256;
        cvt_f32_to_bf16<<<g, 256, 0, stream>>>(Q, Qbf, n4);
        cvt_f32_to_bf16<<<g, 256, 0, stream>>>(K, Kbf, n4);
        cvt_f32_to_bf16<<<g, 256, 0, stream>>>(V, Vbf, n4);
        int w4 = (int)(DD / 4), gw = (w4 + 255) / 256;
        cvt_f32_to_bf16<<<gw, 256, 0, stream>>>(Wq, Wqb, w4);
        cvt_f32_to_bf16<<<gw, 256, 0, stream>>>(Wk, Wkb, w4);
        cvt_f32_to_bf16<<<gw, 256, 0, stream>>>(Wv, Wvb, w4);
        cvt_f32_to_bf16<<<gw, 256, 0, stream>>>(Wo, Wob, w4);
    }

    // 2) projections: Y = X @ W^T + b   (M=4096, N=768, K=768)
    {
        dim3 grid((D_DIM + 127) / 128, S_LEN / 128, 1);
        gemm_bf16_wmma<<<grid, 256, 0, stream>>>(
            Qbf, D_DIM, 0, Wqb, D_DIM, 0, bq, Yq, nullptr, D_DIM, 0,
            S_LEN, D_DIM, D_DIM, 1.0f);
        gemm_bf16_wmma<<<grid, 256, 0, stream>>>(
            Kbf, D_DIM, 0, Wkb, D_DIM, 0, bk, Yk, nullptr, D_DIM, 0,
            S_LEN, D_DIM, D_DIM, 1.0f);
        gemm_bf16_wmma<<<grid, 256, 0, stream>>>(
            Vbf, D_DIM, 0, Wvb, D_DIM, 0, bv, Yv, nullptr, D_DIM, 0,
            S_LEN, D_DIM, D_DIM, 1.0f);
    }

    // 3) V transpose for PV
    {
        int n = (int)SD, g = (n + 255) / 256;
        transpose_v<<<g, 256, 0, stream>>>(Yv, Vt);
    }

    // 4) scores = q k^T / sqrt(DK), per head (z-batched) -> attn (fp32)
    {
        dim3 grid(S_LEN / 128, S_LEN / 128, H_NUM);
        gemm_bf16_wmma<<<grid, 256, 0, stream>>>(
            Yq, D_DIM, DK_DIM,            // A: head h columns [h*64, h*64+64)
            Yk, D_DIM, DK_DIM,            // B: same layout (A*B^T)
            nullptr,
            nullptr, attn, S_LEN, (long)S_LEN * S_LEN,
            S_LEN, S_LEN, DK_DIM, 0.125f);  // 1/sqrt(64)
    }

    // 5) softmax rows -> attn (fp32, in place) + Pbf (bf16)
    softmax_rows<<<H_NUM * S_LEN, 256, 0, stream>>>(attn, Pbf);

    // 6) ctx = P @ V  per head  (M=S, N=DK, K=S; B = Vt is N x K)
    {
        dim3 grid(1, S_LEN / 128, H_NUM);
        gemm_bf16_wmma<<<grid, 256, 0, stream>>>(
            Pbf, S_LEN, (long)S_LEN * S_LEN,
            Vt,  S_LEN, (long)DK_DIM * S_LEN,
            nullptr,
            Ctx, nullptr, D_DIM, DK_DIM,
            S_LEN, DK_DIM, S_LEN, 1.0f);
    }

    // 7) out = Ctx @ Wo^T + bo  -> fp32 d_out
    {
        dim3 grid((D_DIM + 127) / 128, S_LEN / 128, 1);
        gemm_bf16_wmma<<<grid, 256, 0, stream>>>(
            Ctx, D_DIM, 0, Wob, D_DIM, 0, bo, nullptr, out, D_DIM, 0,
            S_LEN, D_DIM, D_DIM, 1.0f);
    }
}
